// LinearTransformerBlock_10247791968597
// MI455X (gfx1250) — matrix operand order repaired and checked by reading, not compiled, not run on hardware
//
#include <hip/hip_runtime.h>
#include <hip/hip_bf16.h>
#include <math.h>

// ---------------------------------------------------------------------------
// LinearTransformerBlock for MI455X (gfx1250, wave32).
//  - 6 large GEMMs run on v_wmma_f32_16x16x32_f16 (f16 in, f32 accum).
//  - GEMM staging uses CDNA5 global_load_async_to_lds_b128 (ASYNCcnt) with
//    double-buffered LDS: tile t+1 streams in while tile t is on the WMMA pipe.
//  - attention kv / q@kv kept in f32 VALU (1e-15 epsilon path is
//    precision-sensitive and compute-trivial: ~3 GFLOP of ~350 GFLOP).
//  - all elementwise work fused into GEMM epilogues.
// Workspace requirement ~400 MB (intermediates + f16 weight copies).
// ---------------------------------------------------------------------------

typedef _Float16 h16;
typedef __attribute__((ext_vector_type(16))) _Float16 v16h;
typedef __attribute__((ext_vector_type(8)))  float    v8f;

#define DIM      1152
#define HEADS    16
#define HEAD_DIM 72
#define HID      4608
#define BATCH    2
#define SEQ      4096
#define MROWS    (BATCH * SEQ)     // 8192 rows in every GEMM
#define KV_STRIDE 80               // padded e-stride for kv accumulator

enum { EPI_H = 0, EPI_RELU_H = 1, EPI_SILU_H = 2, EPI_GATERES_F = 3 };

// CDNA5 async global -> LDS copy, 16 bytes per lane, tracked by ASYNCcnt.
// LDS dest address = low 32 bits of the generic pointer (ISA 10.2: the LDS
// aperture offset is taken from the low address bits).
#define ASYNC_LDS_B128(ldsoff_u32, gptr)                                  \
    asm volatile("global_load_async_to_lds_b128 %0, %1, off"              \
                 :: "v"(ldsoff_u32), "v"(gptr) : "memory")
#define WAIT_ASYNC_3() asm volatile("s_wait_asynccnt 0x3" ::: "memory")
#define WAIT_ASYNC_0() asm volatile("s_wait_asynccnt 0x0" ::: "memory")

// ---------------------------------------------------------------------------
// Tiled WMMA GEMM:  out = epilogue(A(MxK,f16) @ W(NxK,f16)^T + bias)
// Block: 256 threads = 8 waves in a 4x2 grid; tile 128x64, BK=32.
// Each wave owns a 32x32 sub-tile = 2x2 fragments of 16x16.
// Assumes M%128==0, N%64==0, K%32==0 (true for all uses here).
// ---------------------------------------------------------------------------
template <int EPI>
__global__ __launch_bounds__(256)
void gemm_wmma_f16(const h16* __restrict__ A, const h16* __restrict__ W,
                   const float* __restrict__ bias, void* __restrict__ outv,
                   const float* __restrict__ res, const float* __restrict__ mods,
                   int M, int N, int K, int gidx)
{
    constexpr int BM = 128, BN = 64, BK = 32, LDA = 40;  // LDA pad keeps 16B align
    __shared__ h16 As[2][BM * LDA];   // double-buffered
    __shared__ h16 Bs[2][BN * LDA];

    const int tid  = threadIdx.x;
    const int lane = tid & 31;
    const int wid  = tid >> 5;
    const int wm   = wid >> 1;        // 0..3
    const int wn   = wid & 1;         // 0..1
    const int m0   = blockIdx.y * BM;
    const int n0   = blockIdx.x * BN;

    // Per-thread staging slots (fixed across tiles):
    //   A tile: 128x32 halves = 512 x 16B chunks -> 2 chunks/thread
    //   W tile:  64x32 halves = 256 x 16B chunks -> 1 chunk/thread
    const int rA0 = (tid * 2)     >> 2, koA0 = ((tid * 2)     & 3) * 8;
    const int rA1 = (tid * 2 + 1) >> 2, koA1 = ((tid * 2 + 1) & 3) * 8;
    const int rB  = tid >> 2,           koB  = (tid & 3) * 8;

    auto issue_tile = [&](int kt, int buf) {
        const int k0 = kt * BK;
        ASYNC_LDS_B128((uint32_t)(uintptr_t)&As[buf][rA0 * LDA + koA0],
                       (const void*)(A + (size_t)(m0 + rA0) * K + k0 + koA0));
        ASYNC_LDS_B128((uint32_t)(uintptr_t)&As[buf][rA1 * LDA + koA1],
                       (const void*)(A + (size_t)(m0 + rA1) * K + k0 + koA1));
        ASYNC_LDS_B128((uint32_t)(uintptr_t)&Bs[buf][rB * LDA + koB],
                       (const void*)(W + (size_t)(n0 + rB) * K + k0 + koB));
    };

    v8f acc[2][2] = {};
    const int nK = K / BK;

    issue_tile(0, 0);                       // prologue: tile 0 in flight
    for (int kt = 0; kt < nK; ++kt) {
        const int buf = kt & 1;
        if (kt + 1 < nK) {
            issue_tile(kt + 1, buf ^ 1);    // stream next tile into other buffer
            // pull tile kt+2 toward L2 while LDS fills
            __builtin_prefetch(A + (size_t)(m0 + rA0) * K + (kt + 2) * BK, 0, 1);
            __builtin_prefetch(W + (size_t)(n0 + rB)  * K + (kt + 2) * BK, 0, 1);
            WAIT_ASYNC_3();                 // async done in order: tile kt landed
        } else {
            WAIT_ASYNC_0();
        }
        __syncthreads();                    // every wave's tile-kt data visible

        // ---- build fragments per ISA 7.12.2 layouts ----
        v16h af[2], bf[2];
        const int arow = lane & 15;
        const int akk  = (lane >> 4) * 8;   // lanes<16: K0-7/16-23; else K8-15/24-31
#pragma unroll
        for (int i = 0; i < 2; ++i) {
            const h16* p = &As[buf][(wm * 32 + i * 16 + arow) * LDA + akk];
            float4* d = reinterpret_cast<float4*>(&af[i]);
            d[0] = *reinterpret_cast<const float4*>(p);
            d[1] = *reinterpret_cast<const float4*>(p + 16);
        }
        const int bkk = (lane >> 4) * 8;    // lanes<16: K0-7/16-23; else K8-15/24-31
#pragma unroll
        for (int j = 0; j < 2; ++j) {
            const h16* p = &Bs[buf][(wn * 32 + j * 16 + (lane & 15)) * LDA + bkk];
            float4* d = reinterpret_cast<float4*>(&bf[j]);
            d[0] = *reinterpret_cast<const float4*>(p);
            d[1] = *reinterpret_cast<const float4*>(p + 16);
        }

#pragma unroll
        for (int i = 0; i < 2; ++i)
#pragma unroll
            for (int j = 0; j < 2; ++j)
                acc[i][j] = __builtin_amdgcn_wmma_f32_16x16x32_f16(
                    false, af[i], false, bf[j], (short)0, acc[i][j], false, false);
        __syncthreads();                    // buf may now be refilled next iter
    }

    // ---- fused epilogue.  C layout: VGPR r -> row r (lanes 0-15) / r+8 (16-31),
    //      col = lane&15 (ISA 7.12.2). ----
#pragma unroll
    for (int i = 0; i < 2; ++i) {
#pragma unroll
        for (int j = 0; j < 2; ++j) {
            const int col  = n0 + wn * 32 + j * 16 + (lane & 15);
            const float bv = bias ? bias[col] : 0.0f;
#pragma unroll
            for (int r = 0; r < 8; ++r) {
                const int row = m0 + wm * 32 + i * 16 + r + ((lane >> 4) << 3);
                const size_t idx = (size_t)row * N + col;
                float v = acc[i][j][r] + bv;
                if constexpr (EPI == EPI_RELU_H) {
                    ((h16*)outv)[idx] = (h16)fmaxf(v, 0.0f);
                } else if constexpr (EPI == EPI_SILU_H) {
                    ((h16*)outv)[idx] = (h16)(v / (1.0f + __expf(-v)));
                } else if constexpr (EPI == EPI_GATERES_F) {
                    const int b = row / SEQ;
                    ((float*)outv)[idx] = res[idx] + mods[(b * 6 + gidx) * DIM + col] * v;
                } else {
                    ((h16*)outv)[idx] = (h16)v;
                }
            }
        }
    }
}

// ---------------------------------------------------------------------------
// mods[b,i,d] = scale_shift_table[i,d] + temb[b, i*D+d]
// ---------------------------------------------------------------------------
__global__ void mods_kernel(const float* __restrict__ sst,
                            const float* __restrict__ temb,
                            float* __restrict__ mods)
{
    const int i = blockIdx.x * blockDim.x + threadIdx.x;
    if (i < BATCH * 6 * DIM) mods[i] = sst[i % (6 * DIM)] + temb[i];
}

// ---------------------------------------------------------------------------
// f32 -> f16 weight conversion (one pass over each weight matrix)
// ---------------------------------------------------------------------------
__global__ void f32_to_f16_kernel(const float* __restrict__ in,
                                  h16* __restrict__ out, size_t n)
{
    const size_t i = (size_t)blockIdx.x * blockDim.x + threadIdx.x;
    if (i < n) out[i] = (h16)in[i];
}

// ---------------------------------------------------------------------------
// out[row,:] = rms(x[row,:]) * (1 + mods[b,scaleIdx,:]) + mods[b,shiftIdx,:]
// One 256-thread block per row (D = 1152), f32 in -> f16 out.
// ---------------------------------------------------------------------------
__global__ __launch_bounds__(256)
void rmsmod_kernel(const float* __restrict__ x, const float* __restrict__ mods,
                   h16* __restrict__ out, int shiftIdx, int scaleIdx)
{
    __shared__ float red[256];
    const int row = blockIdx.x;
    const int b   = row / SEQ;
    const float* xr = x + (size_t)row * DIM;

    float ss = 0.0f;
    for (int d = threadIdx.x; d < DIM; d += 256) { const float v = xr[d]; ss += v * v; }
    red[threadIdx.x] = ss;
    __syncthreads();
    for (int s = 128; s > 0; s >>= 1) {
        if (threadIdx.x < s) red[threadIdx.x] += red[threadIdx.x + s];
        __syncthreads();
    }
    const float inv = rsqrtf(red[0] / (float)DIM + 1e-6f);
    const float* sh = mods + (size_t)(b * 6 + shiftIdx) * DIM;
    const float* sc = mods + (size_t)(b * 6 + scaleIdx) * DIM;
    for (int d = threadIdx.x; d < DIM; d += 256)
        out[(size_t)row * DIM + d] = (h16)(xr[d] * inv * (1.0f + sc[d]) + sh[d]);
}

// ---------------------------------------------------------------------------
// kv[b,h,d,e] += sum_n k[b,n,h,d] * v_pad[b,n,h,e]   (v_pad[...,72] = 1)
// split-K over n: grid (SEQ/512, B*H); LDS-staged 64-row chunks; f32 atomics.
// ---------------------------------------------------------------------------
__global__ __launch_bounds__(256)
void kv_partial_kernel(const h16* __restrict__ k, const h16* __restrict__ v,
                       float* __restrict__ kv)
{
    __shared__ float kds[64][HEAD_DIM];
    __shared__ float vds[64][HEAD_DIM + 1];
    const int bh   = blockIdx.y;
    const int b    = bh / HEADS, h = bh % HEADS;
    const int base = blockIdx.x * 512;
    const int tid  = threadIdx.x;

    float regacc[21];
#pragma unroll
    for (int i = 0; i < 21; ++i) regacc[i] = 0.0f;

    for (int sc = 0; sc < 8; ++sc) {
        const int n0 = base + sc * 64;
        for (int idx = tid; idx < 64 * HEAD_DIM; idx += 256) {
            const int nn = idx / HEAD_DIM, d = idx % HEAD_DIM;
            const size_t g = (size_t)(b * SEQ + n0 + nn) * DIM + h * HEAD_DIM + d;
            kds[nn][d] = (float)k[g];
            vds[nn][d] = (float)v[g];
        }
        if (tid < 64) vds[tid][HEAD_DIM] = 1.0f;
        __syncthreads();

        int pi = 0;
        for (int p = tid; p < HEAD_DIM * (HEAD_DIM + 1); p += 256, ++pi) {
            const int d = p / (HEAD_DIM + 1), e = p % (HEAD_DIM + 1);
            float s = 0.0f;
#pragma unroll 4
            for (int nn = 0; nn < 64; ++nn) s += kds[nn][d] * vds[nn][e];
            regacc[pi] += s;
        }
        __syncthreads();
    }
    int pi = 0;
    for (int p = tid; p < HEAD_DIM * (HEAD_DIM + 1); p += 256, ++pi) {
        const int d = p / (HEAD_DIM + 1), e = p % (HEAD_DIM + 1);
        atomicAdd(&kv[(size_t)bh * HEAD_DIM * KV_STRIDE + d * KV_STRIDE + e],
                  regacc[pi]);
    }
}

// ---------------------------------------------------------------------------
// out[b,n,h,e] = (q[b,n,h,:] . kv[b,h,:,e]) / (q . kv[:,72] + 1e-15)
// kv tile cached in LDS; one thread per sequence position.
// ---------------------------------------------------------------------------
__global__ __launch_bounds__(128)
void attn_apply_kernel(const h16* __restrict__ q, const float* __restrict__ kv,
                       h16* __restrict__ out)
{
    __shared__ float kvs[HEAD_DIM * KV_STRIDE];
    const int bh = blockIdx.y;
    const int b  = bh / HEADS, h = bh % HEADS;
    for (int i = threadIdx.x; i < HEAD_DIM * KV_STRIDE; i += 128)
        kvs[i] = kv[(size_t)bh * HEAD_DIM * KV_STRIDE + i];
    __syncthreads();

    const int n = blockIdx.x * 128 + threadIdx.x;
    const size_t base = (size_t)(b * SEQ + n) * DIM + h * HEAD_DIM;

    float qr[HEAD_DIM];
#pragma unroll
    for (int d = 0; d < HEAD_DIM; ++d) qr[d] = (float)q[base + d];

    float den = 0.0f;
#pragma unroll 8
    for (int d = 0; d < HEAD_DIM; ++d) den += qr[d] * kvs[d * KV_STRIDE + HEAD_DIM];
    const float invden = 1.0f / (den + 1e-15f);

    for (int e = 0; e < HEAD_DIM; ++e) {
        float num = 0.0f;
#pragma unroll 8
        for (int d = 0; d < HEAD_DIM; ++d) num += qr[d] * kvs[d * KV_STRIDE + e];
        out[base + e] = (h16)(num * invden);
    }
}

// ---------------------------------------------------------------------------
// Depthwise k=3 conv along n (+bias) on SiLU'd y1 (B,N,2*HID), then GLU:
// out[b,n,c] = conv_c * silu(conv_{c+HID}),   f16 out (B,N,HID).
// ---------------------------------------------------------------------------
__global__ __launch_bounds__(256)
void dwconv_glu_kernel(const h16* __restrict__ y1, const float* __restrict__ Wdw,
                       const float* __restrict__ bdw, h16* __restrict__ out)
{
    const int row = blockIdx.x;
    const int b = row / SEQ, n = row % SEQ;
    for (int c = threadIdx.x; c < HID; c += 256) {
        float xg = bdw[c], gt = bdw[c + HID];
#pragma unroll
        for (int t = 0; t < 3; ++t) {
            const int nn = n + t - 1;
            if (nn >= 0 && nn < SEQ) {
                const size_t rbase = (size_t)(b * SEQ + nn) * (2 * HID);
                xg += (float)y1[rbase + c]       * Wdw[c * 3 + t];
                gt += (float)y1[rbase + c + HID] * Wdw[(c + HID) * 3 + t];
            }
        }
        out[(size_t)row * HID + c] = (h16)(xg * (gt / (1.0f + __expf(-gt))));
    }
}

// ---------------------------------------------------------------------------
extern "C" void kernel_launch(void* const* d_in, const int* in_sizes, int n_in,
                              void* d_out, int out_size, void* d_ws, size_t ws_size,
                              hipStream_t stream)
{
    (void)in_sizes; (void)n_in; (void)out_size; (void)ws_size;

    const float* hidden = (const float*)d_in[0];
    const float* temb   = (const float*)d_in[1];
    const float* sst    = (const float*)d_in[2];
    const float* Wq  = (const float*)d_in[3];  const float* bq   = (const float*)d_in[4];
    const float* Wk  = (const float*)d_in[5];  const float* bk   = (const float*)d_in[6];
    const float* Wv  = (const float*)d_in[7];  const float* bv   = (const float*)d_in[8];
    const float* Wo  = (const float*)d_in[9];  const float* bo   = (const float*)d_in[10];
    const float* Winv = (const float*)d_in[11]; const float* binv = (const float*)d_in[12];
    const float* Wdw  = (const float*)d_in[13]; const float* bdw  = (const float*)d_in[14];
    const float* Wpt  = (const float*)d_in[15];

    // bump allocator on workspace (256B aligned)
    char* ws = (char*)d_ws;
    size_t off = 0;
    auto alloc = [&](size_t bytes) -> void* {
        void* p = ws + off;
        off = (off + bytes + 255) & ~(size_t)255;
        return p;
    };

    float* modsb = (float*)alloc((size_t)BATCH * 6 * DIM * 4);
    h16* wq_h  = (h16*)alloc((size_t)DIM * DIM * 2);
    h16* wk_h  = (h16*)alloc((size_t)DIM * DIM * 2);
    h16* wv_h  = (h16*)alloc((size_t)DIM * DIM * 2);
    h16* wo_h  = (h16*)alloc((size_t)DIM * DIM * 2);
    h16* winv_h = (h16*)alloc((size_t)2 * HID * DIM * 2);
    h16* wpt_h  = (h16*)alloc((size_t)DIM * HID * 2);
    h16* x_h    = (h16*)alloc((size_t)MROWS * DIM * 2);
    h16* q_h    = (h16*)alloc((size_t)MROWS * DIM * 2);
    h16* k_h    = (h16*)alloc((size_t)MROWS * DIM * 2);
    h16* v_h    = (h16*)alloc((size_t)MROWS * DIM * 2);
    h16* attn_h = (h16*)alloc((size_t)MROWS * DIM * 2);
    float* kvb  = (float*)alloc((size_t)BATCH * HEADS * HEAD_DIM * KV_STRIDE * 4);
    float* hid2 = (float*)alloc((size_t)MROWS * DIM * 4);
    h16* y_h    = (h16*)alloc((size_t)MROWS * DIM * 2);
    h16* y1_h   = (h16*)alloc((size_t)MROWS * 2 * HID * 2);
    h16* yg_h   = (h16*)alloc((size_t)MROWS * HID * 2);

    // 0) modulation vectors
    mods_kernel<<<(BATCH * 6 * DIM + 255) / 256, 256, 0, stream>>>(sst, temb, modsb);

    // 1) weights -> f16 (stay resident in the 192MB L2 across the GEMMs)
    auto conv = [&](const float* src, h16* dst, size_t n) {
        f32_to_f16_kernel<<<(unsigned)((n + 255) / 256), 256, 0, stream>>>(src, dst, n);
    };
    conv(Wq, wq_h, (size_t)DIM * DIM);
    conv(Wk, wk_h, (size_t)DIM * DIM);
    conv(Wv, wv_h, (size_t)DIM * DIM);
    conv(Wo, wo_h, (size_t)DIM * DIM);
    conv(Winv, winv_h, (size_t)2 * HID * DIM);
    conv(Wpt, wpt_h, (size_t)DIM * HID);

    // 2) x = rms(hidden)*(1+scale_msa)+shift_msa    (shift=0, scale=1)
    rmsmod_kernel<<<MROWS, 256, 0, stream>>>(hidden, modsb, x_h, 0, 1);

    // 3) q/k/v projections (WMMA GEMMs), ReLU fused for q,k
    dim3 gQKV(DIM / 64, MROWS / 128);
    gemm_wmma_f16<EPI_RELU_H><<<gQKV, 256, 0, stream>>>(
        x_h, wq_h, bq, q_h, nullptr, nullptr, MROWS, DIM, DIM, 0);
    gemm_wmma_f16<EPI_RELU_H><<<gQKV, 256, 0, stream>>>(
        x_h, wk_h, bk, k_h, nullptr, nullptr, MROWS, DIM, DIM, 0);
    gemm_wmma_f16<EPI_H><<<gQKV, 256, 0, stream>>>(
        x_h, wv_h, bv, v_h, nullptr, nullptr, MROWS, DIM, DIM, 0);

    // 4) kv = sum_n k^T v_pad   (split-K, f32 atomics)
    hipMemsetAsync(kvb, 0, (size_t)BATCH * HEADS * HEAD_DIM * KV_STRIDE * 4, stream);
    kv_partial_kernel<<<dim3(SEQ / 512, BATCH * HEADS), 256, 0, stream>>>(k_h, v_h, kvb);

    // 5) out = (q @ kv) num/den normalize
    attn_apply_kernel<<<dim3(SEQ / 128, BATCH * HEADS), 128, 0, stream>>>(q_h, kvb, attn_h);

    // 6) hidden2 = gate_msa * (attn @ Wo^T + bo) + hidden  (fused f32 epilogue)
    gemm_wmma_f16<EPI_GATERES_F><<<gQKV, 256, 0, stream>>>(
        attn_h, wo_h, bo, hid2, hidden, modsb, MROWS, DIM, DIM, 2);

    // 7) y = rms(hidden2)*(1+scale_mlp)+shift_mlp   (shift=3, scale=4)
    rmsmod_kernel<<<MROWS, 256, 0, stream>>>(hid2, modsb, y_h, 3, 4);

    // 8) y1 = silu(y @ Winv^T + binv)   (M=8192, N=9216, K=1152)
    gemm_wmma_f16<EPI_SILU_H><<<dim3(2 * HID / 64, MROWS / 128), 256, 0, stream>>>(
        y_h, winv_h, binv, y1_h, nullptr, nullptr, MROWS, 2 * HID, DIM, 0);

    // 9) depthwise conv k=3 + GLU
    dwconv_glu_kernel<<<MROWS, 256, 0, stream>>>(y1_h, Wdw, bdw, yg_h);

    // 10) out = hidden2 + gate_mlp * (yg @ Wpt^T)   (K=4608, no bias)
    gemm_wmma_f16<EPI_GATERES_F><<<dim3(DIM / 64, MROWS / 128), 256, 0, stream>>>(
        yg_h, wpt_h, nullptr, (float*)d_out, hid2, modsb, MROWS, DIM, HID, 5);
}
